// AttnBlock_87737591922720
// MI455X (gfx1250) — compile-verified, matrix operand order checked by reading, not run on hardware
//
#include <hip/hip_runtime.h>
#include <hip/hip_bf16.h>

typedef __attribute__((ext_vector_type(16))) _Float16 v16h;
typedef __attribute__((ext_vector_type(8)))  _Float16 v8h;
typedef __attribute__((ext_vector_type(4)))  _Float16 v4h;
typedef __attribute__((ext_vector_type(2)))  _Float16 v2h;
typedef __attribute__((ext_vector_type(8)))  float    v8f;
typedef __attribute__((ext_vector_type(4)))  float    v4f;
typedef __attribute__((ext_vector_type(4)))  int      v4i;

#define BATCH 8
#define CDIM  256
#define HWDIM 4096
#define NGRP  4
#define TSTR  264     // padded row stride (halfs) for transposed Q/K/O buffers

// ---- feature detection -------------------------------------------------
#if defined(__has_builtin)
#  if __has_builtin(__builtin_amdgcn_global_load_async_to_lds_b128)
#    define HAVE_ASYNC_B128 1
#  endif
#  if __has_builtin(__builtin_amdgcn_s_wait_asynccnt)
#    define HAVE_WAIT_ASYNC 1
#  endif
#endif

// Builtin signature (confirmed via clang diagnostic):
// (v4i addrspace(1)*, v4i addrspace(3)*, i32 offset, i32 cpol)
typedef __attribute__((address_space(1))) v4i* gv4i_ptr;
typedef __attribute__((address_space(3))) v4i* lv4i_ptr;

__device__ __forceinline__ void async_copy_b128(const _Float16* gsrc, _Float16* ldst) {
#ifdef HAVE_ASYNC_B128
    __builtin_amdgcn_global_load_async_to_lds_b128(
        (gv4i_ptr)(void*)gsrc, (lv4i_ptr)(void*)ldst, 0, 0);
#else
    unsigned lofs = (unsigned)(unsigned long long)ldst;
    asm volatile("global_load_async_to_lds_b128 %0, %1, off"
                 :: "v"(lofs), "v"((unsigned long long)gsrc) : "memory");
#endif
}

__device__ __forceinline__ void wait_async0() {
#ifdef HAVE_WAIT_ASYNC
    __builtin_amdgcn_s_wait_asynccnt(0);
#else
    asm volatile("s_wait_asynccnt 0x0" ::: "memory");
#endif
}
// wait until <=16 outstanding: async loads complete in-order, so this
// guarantees the previously-issued tile (16 ops/thread) has landed.
__device__ __forceinline__ void wait_async16() {
#ifdef HAVE_WAIT_ASYNC
    __builtin_amdgcn_s_wait_asynccnt(16);
#else
    asm volatile("s_wait_asynccnt 0x10" ::: "memory");
#endif
}

// ---- WMMA helpers ------------------------------------------------------
__device__ __forceinline__ v8f wmma_f16(v16h a, v16h b, v8f c) {
    return __builtin_amdgcn_wmma_f32_16x16x32_f16(
        false, a, false, b, (short)0, c, false, false);
}

// Load one 16x32 f16 A/B fragment from a row-major LDS tile.
__device__ __forceinline__ v16h load_frag(const _Float16* base, int row, int stride, int kofs) {
    int lane = threadIdx.x & 31;
    int m    = lane & 15;
    int hf   = lane >> 4;
    const _Float16* p = base + (size_t)(row + m) * stride + kofs + (hf << 3);
    v8h lo = *(const v8h*)(p);
    v8h hi = *(const v8h*)(p + 16);
    return __builtin_shufflevector(lo, hi, 0,1,2,3,4,5,6,7,8,9,10,11,12,13,14,15);
}

// ---- Kernel 1: GroupNorm statistics ------------------------------------
__global__ __launch_bounds__(256)
void gn_stats_kernel(const float* __restrict__ x, const float* __restrict__ y,
                     float* __restrict__ stats) {
    const int bg  = blockIdx.x;
    const int src = blockIdx.y;
    const v4f* p4 = (const v4f*)((src == 0 ? x : y) + (size_t)bg * 64 * HWDIM);
    const int n4 = 64 * HWDIM / 4;

    __shared__ float ssum[256];
    __shared__ float ssq[256];
    float s = 0.f, q = 0.f;
    for (int i = threadIdx.x; i < n4; i += 256) {
        v4f v = p4[i];
        s += v.x + v.y + v.z + v.w;
        q += v.x * v.x + v.y * v.y + v.z * v.z + v.w * v.w;
    }
    ssum[threadIdx.x] = s; ssq[threadIdx.x] = q;
    __syncthreads();
    for (int off = 128; off > 0; off >>= 1) {
        if (threadIdx.x < off) {
            ssum[threadIdx.x] += ssum[threadIdx.x + off];
            ssq[threadIdx.x]  += ssq[threadIdx.x + off];
        }
        __syncthreads();
    }
    if (threadIdx.x == 0) {
        const float inv_n = 1.0f / (float)(64 * HWDIM);
        float mean = ssum[0] * inv_n;
        float var  = ssq[0] * inv_n - mean * mean;
        stats[(size_t)(src * 32 + bg) * 2 + 0] = mean;
        stats[(size_t)(src * 32 + bg) * 2 + 1] = rsqrtf(var + 1e-6f);
    }
}

// ---- Kernel 2: fused GroupNorm + QKV 1x1 conv --------------------------
// Q,K written transposed+padded [b][n][TSTR] (contiguous b128 stores);
// V written [b][c][n].  grid: (N/64, C/64, B*3)  block: 128
__global__ __launch_bounds__(128)
void qkv_kernel(const float* __restrict__ x, const float* __restrict__ y,
                const float* __restrict__ wq, const float* __restrict__ bq,
                const float* __restrict__ wk, const float* __restrict__ bk,
                const float* __restrict__ wv, const float* __restrict__ bv,
                const float* __restrict__ gamma, const float* __restrict__ beta,
                const float* __restrict__ stats,
                _Float16* __restrict__ Qb, _Float16* __restrict__ Kb,
                _Float16* __restrict__ Vb) {
    const int nb = blockIdx.x * 64;
    const int mb = blockIdx.y * 64;
    const int b  = blockIdx.z / 3;
    const int proj = blockIdx.z % 3;

    const float* src  = (proj == 0) ? x : y;
    const int statsrc = (proj == 0) ? 0 : 1;
    const float* W    = (proj == 0) ? wq : (proj == 1 ? wk : wv);
    const float* bias = (proj == 0) ? bq : (proj == 1 ? bk : bv);
    _Float16*  dst    = (proj == 0) ? Qb : (proj == 1 ? Kb : Vb);

    __shared__ _Float16 Wt[64 * 40];
    __shared__ _Float16 Xt[64 * 40];
    __shared__ float biasS[64];
    __shared__ float meanS[NGRP], rstdS[NGRP];

    const int tid = threadIdx.x;
    if (tid < 64) biasS[tid] = bias[mb + tid];
    if (tid < NGRP) {
        meanS[tid] = stats[(size_t)(statsrc * 32 + b * 4 + tid) * 2 + 0];
        rstdS[tid] = stats[(size_t)(statsrc * 32 + b * 4 + tid) * 2 + 1];
    }

    const int w = tid >> 5;
    const int mw = w >> 1, nw = w & 1;
    const int lane = tid & 31, ln = lane & 15, hf = lane >> 4;

    v8f acc[2][2];
    #pragma unroll
    for (int i = 0; i < 2; ++i)
        #pragma unroll
        for (int j = 0; j < 2; ++j)
            #pragma unroll
            for (int r = 0; r < 8; ++r) acc[i][j][r] = 0.f;

    const float* srcb = src + (size_t)b * CDIM * HWDIM;

    for (int ks = 0; ks < 8; ++ks) {
        const int cb = ks * 32;
        __syncthreads();
        for (int i = tid; i < 512; i += 128) {
            int r = i >> 3, c4 = (i & 7) * 4;
            v4f wv4 = *(const v4f*)(W + (size_t)(mb + r) * CDIM + cb + c4);
            v4h h; h.x = (_Float16)wv4.x; h.y = (_Float16)wv4.y;
                   h.z = (_Float16)wv4.z; h.w = (_Float16)wv4.w;
            *(v4h*)(Wt + r * 40 + c4) = h;
        }
        for (int i = tid; i < 1024; i += 128) {
            int c = (i >> 6) * 2, n = i & 63;
            int cin = cb + c;
            int g = cin >> 6;
            float v0 = srcb[(size_t)cin * HWDIM + nb + n];
            float v1 = srcb[(size_t)(cin + 1) * HWDIM + nb + n];
            v0 = (v0 - meanS[g]) * rstdS[g] * gamma[cin]     + beta[cin];
            v1 = (v1 - meanS[g]) * rstdS[g] * gamma[cin + 1] + beta[cin + 1];
            v2h p; p.x = (_Float16)v0; p.y = (_Float16)v1;
            *(v2h*)(Xt + n * 40 + c) = p;
        }
        __syncthreads();
        v16h a0 = load_frag(Wt, mw * 32,      40, 0);
        v16h a1 = load_frag(Wt, mw * 32 + 16, 40, 0);
        v16h b0 = load_frag(Xt, nw * 32,      40, 0);
        v16h b1 = load_frag(Xt, nw * 32 + 16, 40, 0);
        acc[0][0] = wmma_f16(a0, b0, acc[0][0]);
        acc[0][1] = wmma_f16(a0, b1, acc[0][1]);
        acc[1][0] = wmma_f16(a1, b0, acc[1][0]);
        acc[1][1] = wmma_f16(a1, b1, acc[1][1]);
    }

    const float scale = (proj == 0) ? 0.0625f : 1.0f;   // C^-0.5 folded into Q
    if (proj < 2) {
        // transposed padded layout [n][TSTR]: one b128 store per fragment
        _Float16* dstb = dst + (size_t)b * HWDIM * TSTR;
        #pragma unroll
        for (int mf = 0; mf < 2; ++mf)
            #pragma unroll
            for (int nf = 0; nf < 2; ++nf) {
                const int col  = nb + nw * 32 + nf * 16 + ln;
                const int rbase = mw * 32 + mf * 16 + hf * 8;
                v8h o;
                #pragma unroll
                for (int r = 0; r < 8; ++r)
                    o[r] = (_Float16)((acc[mf][nf][r] + biasS[rbase + r]) * scale);
                *(v8h*)(dstb + (size_t)col * TSTR + mb + rbase) = o;
            }
    } else {
        _Float16* dstb = dst + (size_t)b * CDIM * HWDIM;
        #pragma unroll
        for (int mf = 0; mf < 2; ++mf)
            #pragma unroll
            for (int nf = 0; nf < 2; ++nf) {
                const int col = nb + nw * 32 + nf * 16 + ln;
                #pragma unroll
                for (int r = 0; r < 8; ++r) {
                    int rowl = mw * 32 + mf * 16 + r + hf * 8;
                    dstb[(size_t)(mb + rowl) * HWDIM + col] =
                        (_Float16)(acc[mf][nf][r] + biasS[rowl]);
                }
            }
    }
}

// ---- Kernel 3: flash attention -----------------------------------------
// grid: (N/64, B)  block: 256 (8 waves).
// All tiles staged by async global->LDS DMA; K/V double-buffered with
// one-tile-ahead issue (s_wait_asynccnt 16 pipelining).
#define QT_STR 264
#define VT_STR 72
#define SB_STR 68
#define PQ_STR 72
#define VT_TILE_B (256 * VT_STR * 2)     // 36864 (also covers Qt: 64*264*2=33792)
#define KT_TILE_B (64 * QT_STR * 2)      // 33792
#define OFF_VT0 0
#define OFF_KT0 (VT_TILE_B)
#define OFF_VT1 (VT_TILE_B + KT_TILE_B)
#define OFF_KT1 (2 * VT_TILE_B + KT_TILE_B)
#define OFF_SB  (2 * VT_TILE_B + 2 * KT_TILE_B)
#define OFF_PQ  (OFF_SB + 64 * SB_STR * 4)
#define OFF_ST  (OFF_PQ + 64 * PQ_STR * 2)
#define ATTN_SMEM_BYTES (OFF_ST + 3 * 64 * 4)   // 168704

__device__ __forceinline__ void attn_issue_tile(const _Float16* Kg, const _Float16* Vg,
                                                int k0, _Float16* Kt, _Float16* Vt, int tid) {
    // V tile [256 c][64 k] from Vg [c][4096]   (8 ops/thread)
    for (int i = tid; i < 2048; i += 256) {
        int c = i >> 3, k8 = (i & 7) * 8;
        async_copy_b128(Vg + (size_t)c * HWDIM + k0 + k8, Vt + (size_t)c * VT_STR + k8);
    }
    // K^T tile [64 k][256 c] from Kg [n][TSTR] (8 ops/thread)
    for (int i = tid; i < 2048; i += 256) {
        int k = i >> 5, c8 = (i & 31) * 8;
        async_copy_b128(Kg + (size_t)(k0 + k) * TSTR + c8, Kt + (size_t)k * QT_STR + c8);
    }
}

__global__ __launch_bounds__(256)
void attn_kernel(const _Float16* __restrict__ Qb, const _Float16* __restrict__ Kb,
                 const _Float16* __restrict__ Vb, _Float16* __restrict__ Ob) {
    extern __shared__ char smem[];
    _Float16* VtB0 = (_Float16*)(smem + OFF_VT0);
    _Float16* VtB1 = (_Float16*)(smem + OFF_VT1);
    _Float16* KtB0 = (_Float16*)(smem + OFF_KT0);
    _Float16* KtB1 = (_Float16*)(smem + OFF_KT1);
    _Float16* Qt   = (_Float16*)(smem + OFF_VT0);   // overlays VtB0 during init
    float*    Sb   = (float*)(smem + OFF_SB);
    _Float16* Pq   = (_Float16*)(smem + OFF_PQ);
    float* mstate  = (float*)(smem + OFF_ST);
    float* lstate  = mstate + 64;
    float* alpha   = lstate + 64;

    const int tid = threadIdx.x;
    const int b   = blockIdx.y;
    const int qb0 = blockIdx.x * 64;
    const _Float16* Qg = Qb + (size_t)b * HWDIM * TSTR;  // [n][TSTR]
    const _Float16* Kg = Kb + (size_t)b * HWDIM * TSTR;  // [n][TSTR]
    const _Float16* Vg = Vb + (size_t)b * CDIM * HWDIM;  // [c][4096]

    if (tid < 64) { mstate[tid] = -1e30f; lstate[tid] = 0.f; }

    // Q^T tile: async identity copy (rows qb0..qb0+63)
    for (int i = tid; i < 2048; i += 256) {
        int q = i >> 5, c8 = (i & 31) * 8;
        async_copy_b128(Qg + (size_t)(qb0 + q) * TSTR + c8, Qt + (size_t)q * QT_STR + c8);
    }
    wait_async0();
    __syncthreads();

    const int w = tid >> 5;
    const int lane = tid & 31, ln = lane & 15, hf = lane >> 4;
    const int qf_s  = w >> 1;
    const int kf_s0 = (w & 1) * 2;

    // hoist Q A-fragments into registers (invariant over all key blocks)
    v16h qa[8];
    #pragma unroll
    for (int s = 0; s < 8; ++s) qa[s] = load_frag(Qt, qf_s * 16, QT_STR, s * 32);
    __syncthreads();   // all waves read Qt before tile 0 overwrites it

    // prologue: issue tile 0 into buffer 0
    attn_issue_tile(Kg, Vg, 0, KtB0, VtB0, tid);

    v8f accO[2][4];
    #pragma unroll
    for (int i = 0; i < 2; ++i)
        #pragma unroll
        for (int j = 0; j < 4; ++j)
            #pragma unroll
            for (int r = 0; r < 8; ++r) accO[i][j][r] = 0.f;

    for (int kb = 0; kb < HWDIM / 64; ++kb) {
        __syncthreads();   // all waves done with buf[(kb+1)&1] (used in kb-1)
        if (kb < HWDIM / 64 - 1) {
            attn_issue_tile(Kg, Vg, (kb + 1) * 64,
                            ((kb + 1) & 1) ? KtB1 : KtB0,
                            ((kb + 1) & 1) ? VtB1 : VtB0, tid);
            wait_async16();   // tile kb (older 16 ops) resident
        } else {
            wait_async0();
        }
        __syncthreads();   // tile kb visible to every wave
        const _Float16* Kt = (kb & 1) ? KtB1 : KtB0;
        const _Float16* Vt = (kb & 1) ? VtB1 : VtB0;

        // ---- S = Q^T K ----
        #pragma unroll
        for (int kk = 0; kk < 2; ++kk) {
            const int kf = kf_s0 + kk;
            v8f acc;
            #pragma unroll
            for (int r = 0; r < 8; ++r) acc[r] = 0.f;
            #pragma unroll
            for (int s = 0; s < 8; ++s) {
                v16h bb = load_frag(Kt, kf * 16, QT_STR, s * 32);
                acc = wmma_f16(qa[s], bb, acc);
            }
            #pragma unroll
            for (int r = 0; r < 8; ++r)
                Sb[(qf_s * 16 + r + hf * 8) * SB_STR + kf * 16 + ln] = acc[r];
        }
        __syncthreads();

        // ---- online softmax ----
        if (tid < 64) {
            float mold = mstate[tid];
            float mmax = mold;
            const float* srow = Sb + tid * SB_STR;
            for (int k = 0; k < 64; ++k) mmax = fmaxf(mmax, srow[k]);
            float al = __expf(mold - mmax);
            float sum = 0.f;
            for (int k = 0; k < 64; ++k) {
                float pv = __expf(srow[k] - mmax);
                sum += pv;
                Pq[tid * PQ_STR + k] = (_Float16)pv;
            }
            mstate[tid] = mmax;
            lstate[tid] = lstate[tid] * al + sum;
            alpha[tid]  = al;
        }
        __syncthreads();

        // ---- rescale running O, then O += V * P^T ----
        #pragma unroll
        for (int qf = 0; qf < 4; ++qf) {
            float al = alpha[qf * 16 + ln];
            #pragma unroll
            for (int r = 0; r < 8; ++r) { accO[0][qf][r] *= al; accO[1][qf][r] *= al; }
        }
        #pragma unroll
        for (int s = 0; s < 2; ++s) {
            v16h a0 = load_frag(Vt, w * 32,      VT_STR, s * 32);
            v16h a1 = load_frag(Vt, w * 32 + 16, VT_STR, s * 32);
            #pragma unroll
            for (int qf = 0; qf < 4; ++qf) {
                v16h bb = load_frag(Pq, qf * 16, PQ_STR, s * 32);
                accO[0][qf] = wmma_f16(a0, bb, accO[0][qf]);
                accO[1][qf] = wmma_f16(a1, bb, accO[1][qf]);
            }
        }
    }

    __syncthreads();
    if (tid < 64) lstate[tid] = 1.0f / fmaxf(lstate[tid], 1e-30f);
    __syncthreads();

    // O written transposed+padded [n][TSTR]: one b128 store per fragment
    _Float16* Og = Ob + (size_t)b * HWDIM * TSTR;
    #pragma unroll
    for (int mf = 0; mf < 2; ++mf)
        #pragma unroll
        for (int qf = 0; qf < 4; ++qf) {
            const float inv = lstate[qf * 16 + ln];
            const int col   = qb0 + qf * 16 + ln;
            const int rbase = w * 32 + mf * 16 + hf * 8;
            v8h o;
            #pragma unroll
            for (int r = 0; r < 8; ++r)
                o[r] = (_Float16)(accO[mf][qf][r] * inv);
            *(v8h*)(Og + (size_t)col * TSTR + rbase) = o;
        }
}

// ---- Kernel 4: output 1x1 conv + residual ------------------------------
// O input is [n][TSTR] so activation staging is an async identity copy.
// grid: (N/64, C/64, B)  block: 128
__global__ __launch_bounds__(128)
void proj_kernel(const _Float16* __restrict__ Ob, const float* __restrict__ wp,
                 const float* __restrict__ bp, const float* __restrict__ x,
                 float* __restrict__ out) {
    const int nb = blockIdx.x * 64;
    const int mb = blockIdx.y * 64;
    const int b  = blockIdx.z;

    __shared__ _Float16 Wt[64 * 40];
    __shared__ _Float16 Xt[64 * 40];
    __shared__ float biasS[64];

    const int tid = threadIdx.x;
    if (tid < 64) biasS[tid] = bp[mb + tid];
    const int w = tid >> 5;
    const int mw = w >> 1, nw = w & 1;
    const int lane = tid & 31, ln = lane & 15, hf = lane >> 4;

    v8f acc[2][2];
    #pragma unroll
    for (int i = 0; i < 2; ++i)
        #pragma unroll
        for (int j = 0; j < 2; ++j)
            #pragma unroll
            for (int r = 0; r < 8; ++r) acc[i][j][r] = 0.f;

    const _Float16* Og = Ob + (size_t)b * HWDIM * TSTR;  // [n][TSTR]

    for (int ks = 0; ks < 8; ++ks) {
        const int cb = ks * 32;
        __syncthreads();
        // activation tile [64 n][32 c]: async identity copy (2 ops/thread)
        for (int i = tid; i < 256; i += 128) {
            int n = i >> 2, c8 = (i & 3) * 8;
            async_copy_b128(Og + (size_t)(nb + n) * TSTR + cb + c8,
                            Xt + (size_t)n * 40 + c8);
        }
        // weight tile
        for (int i = tid; i < 512; i += 128) {
            int r = i >> 3, c4 = (i & 7) * 4;
            v4f wv4 = *(const v4f*)(wp + (size_t)(mb + r) * CDIM + cb + c4);
            v4h h; h.x = (_Float16)wv4.x; h.y = (_Float16)wv4.y;
                   h.z = (_Float16)wv4.z; h.w = (_Float16)wv4.w;
            *(v4h*)(Wt + r * 40 + c4) = h;
        }
        wait_async0();
        __syncthreads();
        v16h a0 = load_frag(Wt, mw * 32,      40, 0);
        v16h a1 = load_frag(Wt, mw * 32 + 16, 40, 0);
        v16h b0 = load_frag(Xt, nw * 32,      40, 0);
        v16h b1 = load_frag(Xt, nw * 32 + 16, 40, 0);
        acc[0][0] = wmma_f16(a0, b0, acc[0][0]);
        acc[0][1] = wmma_f16(a0, b1, acc[0][1]);
        acc[1][0] = wmma_f16(a1, b0, acc[1][0]);
        acc[1][1] = wmma_f16(a1, b1, acc[1][1]);
    }

    const float* xb = x + (size_t)b * CDIM * HWDIM;
    float* ob = out + (size_t)b * CDIM * HWDIM;
    #pragma unroll
    for (int mf = 0; mf < 2; ++mf)
        #pragma unroll
        for (int nf = 0; nf < 2; ++nf) {
            const int col = nb + nw * 32 + nf * 16 + ln;
            #pragma unroll
            for (int r = 0; r < 8; ++r) {
                int rowl = mw * 32 + mf * 16 + r + hf * 8;
                size_t idx = (size_t)(mb + rowl) * HWDIM + col;
                ob[idx] = acc[mf][nf][r] + biasS[rowl] + xb[idx];
            }
        }
}

// ---- launch ------------------------------------------------------------
extern "C" void kernel_launch(void* const* d_in, const int* in_sizes, int n_in,
                              void* d_out, int out_size, void* d_ws, size_t ws_size,
                              hipStream_t stream) {
    const float* x     = (const float*)d_in[0];
    const float* y     = (const float*)d_in[1];
    const float* gamma = (const float*)d_in[2];
    const float* beta  = (const float*)d_in[3];
    const float* wq    = (const float*)d_in[4];
    const float* bq    = (const float*)d_in[5];
    const float* wk    = (const float*)d_in[6];
    const float* bk    = (const float*)d_in[7];
    const float* wv    = (const float*)d_in[8];
    const float* bv    = (const float*)d_in[9];
    const float* wp    = (const float*)d_in[10];
    const float* bp    = (const float*)d_in[11];
    float* out = (float*)d_out;

    char* ws = (char*)d_ws;
    const size_t BUF_T = (size_t)BATCH * HWDIM * TSTR;   // transposed padded (Q,K,O)
    const size_t BUF_N = (size_t)BATCH * CDIM * HWDIM;   // channel-major (V)
    float*     stats = (float*)ws;
    _Float16*  Qb  = (_Float16*)(ws + 1024);
    _Float16*  Kb  = Qb + BUF_T;
    _Float16*  Vb  = Kb + BUF_T;
    _Float16*  Obf = Vb + BUF_N;

    gn_stats_kernel<<<dim3(BATCH * NGRP, 2), 256, 0, stream>>>(x, y, stats);

    qkv_kernel<<<dim3(HWDIM / 64, CDIM / 64, BATCH * 3), 128, 0, stream>>>(
        x, y, wq, bq, wk, bk, wv, bv, gamma, beta, stats, Qb, Kb, Vb);

    attn_kernel<<<dim3(HWDIM / 64, BATCH), 256, (size_t)ATTN_SMEM_BYTES, stream>>>(
        Qb, Kb, Vb, Obf);

    proj_kernel<<<dim3(HWDIM / 64, CDIM / 64, BATCH), 128, 0, stream>>>(
        Obf, wp, bp, x, out);
}
